// LRPEBGCN_19035295056434
// MI455X (gfx1250) — matrix operand order repaired and checked by reading, not compiled, not run on hardware
//
#include <hip/hip_runtime.h>
#include <hip/hip_bf16.h>

#define Nn 8192
#define Ee 8192
#define Bb 64
#define FIN 768
#define HID 64
#define XC 832     // HID + FIN
#define OC 128     // OUT + HID

typedef __attribute__((ext_vector_type(16))) _Float16 v16h;
typedef __attribute__((ext_vector_type(8)))  _Float16 v8h;
typedef __attribute__((ext_vector_type(8)))  float    v8f;

__device__ __forceinline__ float sgnf(float x) {
    return (x > 0.f) ? 1.f : ((x < 0.f) ? -1.f : 0.f);
}

// Build a 16-half WMMA fragment from two 16-byte-aligned 8-half LDS runs.
__device__ __forceinline__ v16h frag16(const _Float16* p0, const _Float16* p1) {
    v8h lo = *(const v8h*)p0;
    v8h hi = *(const v8h*)p1;
    return __builtin_shufflevector(lo, hi, 0, 1, 2, 3, 4, 5, 6, 7,
                                   8, 9, 10, 11, 12, 13, 14, 15);
}

// ---------------------------------------------------------------------------
// Small elementwise / scatter kernels
// ---------------------------------------------------------------------------
__global__ void fill_k(float* p, float v, long long n) {
    long long i = (long long)blockIdx.x * blockDim.x + threadIdx.x;
    if (i < n) p[i] = v;
}

__global__ void deg_count1(const int* __restrict__ dst, float* deg, int E) {
    int e = blockIdx.x * blockDim.x + threadIdx.x;
    if (e < E) atomicAdd(&deg[dst[e]], 1.0f);
}

__global__ void deg_count2(const int* __restrict__ dst, const float* __restrict__ ep,
                           float* deg, int E) {
    int e = blockIdx.x * blockDim.x + threadIdx.x;
    if (e < E) atomicAdd(&deg[dst[e]], ep[e]);
}

__global__ void rsqrt_ip(float* p, int n) {
    int i = blockIdx.x * blockDim.x + threadIdx.x;
    if (i < n) p[i] = rsqrtf(p[i]);
}

// out[n,c] = bias[c] + dinv[n]^2 * lin[n,c]
__global__ void node_init64(float* __restrict__ out, const float* __restrict__ lin,
                            const float* __restrict__ dinv, const float* __restrict__ bias,
                            int Ntot) {
    int idx = blockIdx.x * blockDim.x + threadIdx.x;
    if (idx >= Ntot * HID) return;
    int n = idx >> 6, c = idx & 63;
    float di = dinv[n];
    out[idx] = bias[c] + di * di * lin[idx];
}

// out[d,c] += dinv[s]*dinv[d]*w * lin[s,c]
__global__ void edge_agg(const int* __restrict__ src, const int* __restrict__ dst,
                         const float* __restrict__ dinv, const float* __restrict__ ew,
                         const float* __restrict__ lin, float* out, int E) {
    int idx = blockIdx.x * blockDim.x + threadIdx.x;
    if (idx >= E * HID) return;
    int e = idx >> 6, c = idx & 63;
    int s = src[e], d = dst[e];
    float w = ew ? ew[e] : 1.0f;
    atomicAdd(&out[(size_t)d * HID + c], dinv[s] * dinv[d] * w * lin[(size_t)s * HID + c]);
}

// reversed aggregation for A^T: out[s,c] += dinv[s]*dinv[d] * g[d,c]
__global__ void edge_agg_rev(const int* __restrict__ src, const int* __restrict__ dst,
                             const float* __restrict__ dinv, const float* __restrict__ g,
                             float* out, int E) {
    int idx = blockIdx.x * blockDim.x + threadIdx.x;
    if (idx >= E * HID) return;
    int e = idx >> 6, c = idx & 63;
    int s = src[e], d = dst[e];
    atomicAdd(&out[(size_t)s * HID + c], dinv[s] * dinv[d] * g[(size_t)d * HID + c]);
}

// xc = relu(BN(concat(h, x[root[batch]])))
__global__ void xc_build(const float* __restrict__ h, const float* __restrict__ x,
                         const int* __restrict__ batch, const int* __restrict__ root,
                         const float* __restrict__ bw, const float* __restrict__ bb,
                         const float* __restrict__ brm, const float* __restrict__ brv,
                         float* xc, int Ntot) {
    int idx = blockIdx.x * blockDim.x + threadIdx.x;
    if (idx >= Ntot * XC) return;
    int n = idx / XC, c = idx % XC;
    float v = (c < HID) ? h[(size_t)n * HID + c]
                        : x[(size_t)root[batch[n]] * FIN + (c - HID)];
    float scale = bw[c] * rsqrtf(brv[c] + 1e-5f);
    v = (v - brm[c]) * scale + bb[c];
    xc[idx] = v > 0.f ? v : 0.f;
}

__global__ void cnt_count(const int* __restrict__ batch, float* cntf, int Ntot) {
    int n = blockIdx.x * blockDim.x + threadIdx.x;
    if (n < Ntot) atomicAdd(&cntf[batch[n]], 1.0f);
}

// Zsum[b, :] += concat(relu(out_pre[n]), h[root[b]])
__global__ void z_accum(const float* __restrict__ out_pre, const float* __restrict__ h,
                        const int* __restrict__ batch, const int* __restrict__ root,
                        float* Zsum, int Ntot) {
    int idx = blockIdx.x * blockDim.x + threadIdx.x;
    if (idx >= Ntot * OC) return;
    int n = idx >> 7, c = idx & 127;
    int b = batch[n];
    float v = (c < HID) ? fmaxf(out_pre[(size_t)n * HID + c], 0.f)
                        : h[(size_t)root[b] * HID + (c - HID)];
    atomicAdd(&Zsum[b * OC + c], v);
}

// S = safe_divide(R, Z)
__global__ void s_build(const float* __restrict__ Zsum, const float* __restrict__ cntf,
                        const float* __restrict__ R, float* S, int tot) {
    int i = blockIdx.x * blockDim.x + threadIdx.x;
    if (i >= tot) return;
    float z = Zsum[i] / fmaxf(cntf[i >> 7], 1.f);
    float den = z + 1e-6f * sgnf(z);
    if (den == 0.f) den = 1e-12f;
    S[i] = (R[i] / den) * (z != 0.f ? 1.f : 0.f);
}

// grad of mean + split of out_cat
__global__ void bwd_mean_split(const float* __restrict__ out_pre, const float* __restrict__ S,
                               const int* __restrict__ batch, const int* __restrict__ root,
                               const float* __restrict__ cntf,
                               float* g_outpre, float* g_h, int Ntot) {
    int idx = blockIdx.x * blockDim.x + threadIdx.x;
    if (idx >= Ntot * OC) return;
    int n = idx >> 7, c = idx & 127;
    int b = batch[n];
    float g = S[b * OC + c] / fmaxf(cntf[b], 1.f);
    if (c < HID)
        g_outpre[(size_t)n * HID + c] = (out_pre[(size_t)n * HID + c] > 0.f) ? g : 0.f;
    else
        atomicAdd(&g_h[(size_t)root[b] * HID + (c - HID)], g);
}

// conv2 backward: self-loop terms (per node)
__global__ void conv2_bwd_self(const float* __restrict__ g_outpre, const float* __restrict__ hh,
                               const float* __restrict__ dinv2,
                               float* g_hh, float* g_dinv2, int Ntot) {
    int n = blockIdx.x * blockDim.x + threadIdx.x;
    if (n >= Ntot) return;
    float di = dinv2[n], g0 = di * di, sdot = 0.f;
    size_t base = (size_t)n * HID;
    for (int c = 0; c < HID; ++c) {
        float gp = g_outpre[base + c];
        g_hh[base + c] = g0 * gp;
        sdot += gp * hh[base + c];
    }
    g_dinv2[n] = 2.f * di * sdot;
}

// conv2 backward: per-edge terms (64 threads / edge)
__global__ void conv2_bwd_edge(const int* __restrict__ src, const int* __restrict__ dst,
                               const float* __restrict__ g_outpre, const float* __restrict__ hh,
                               const float* __restrict__ dinv2, const float* __restrict__ ep,
                               float* g_ep, float* g_dinv2, float* g_hh) {
    __shared__ float red[64];
    int e = blockIdx.x, t = threadIdx.x;
    int s = src[e], d = dst[e];
    float gp = g_outpre[(size_t)d * HID + t];
    float hv = hh[(size_t)s * HID + t];
    red[t] = gp * hv;
    __syncthreads();
    for (int st = 32; st > 0; st >>= 1) {
        if (t < st) red[t] += red[t + st];
        __syncthreads();
    }
    float ds_ = dinv2[s], dd = dinv2[d], w = ep[e];
    if (t == 0) {
        float gdot = red[0];
        g_ep[e] += gdot * ds_ * dd;
        atomicAdd(&g_dinv2[s], gdot * w * dd);
        atomicAdd(&g_dinv2[d], gdot * w * ds_);
    }
    atomicAdd(&g_hh[(size_t)s * HID + t], ds_ * w * dd * gp);
}

// g_dinv2 -> g_deg2 (in place)
__global__ void dinv2deg(const float* __restrict__ dinv2, float* g_dinv2, int Ntot) {
    int n = blockIdx.x * blockDim.x + threadIdx.x;
    if (n >= Ntot) return;
    float d3 = dinv2[n];
    g_dinv2[n] = -0.5f * d3 * d3 * d3 * g_dinv2[n];
}

__global__ void deg2ep(const int* __restrict__ dst, const float* __restrict__ g_deg2,
                       float* g_ep, int E) {
    int e = blockIdx.x * blockDim.x + threadIdx.x;
    if (e < E) g_ep[e] += g_deg2[dst[e]];
}

// split of xc gradient: BN/relu backward, scatter to g_h / g_x
__global__ void xc_bwd_split(const float* __restrict__ gxc, const float* __restrict__ xc_act,
                             const float* __restrict__ bw, const float* __restrict__ brv,
                             const int* __restrict__ batch, const int* __restrict__ root,
                             float* g_h, float* g_x, int Ntot) {
    int idx = blockIdx.x * blockDim.x + threadIdx.x;
    if (idx >= Ntot * XC) return;
    int n = idx / XC, c = idx % XC;
    float gv = gxc[idx] * (xc_act[idx] > 0.f ? 1.f : 0.f) * bw[c] * rsqrtf(brv[c] + 1e-5f);
    if (c < HID)
        g_h[(size_t)n * HID + c] += gv;
    else
        atomicAdd(&g_x[(size_t)root[batch[n]] * FIN + (c - HID)], gv);
}

// gA[n,c] = dinv1[n]^2 * g_h[n,c]
__global__ void node_scale64(const float* __restrict__ g_h, const float* __restrict__ dinv,
                             float* gA, int Ntot) {
    int idx = blockIdx.x * blockDim.x + threadIdx.x;
    if (idx >= Ntot * HID) return;
    int n = idx >> 6;
    float di = dinv[n];
    gA[idx] = di * di * g_h[idx];
}

// ---------------------------------------------------------------------------
// Generic WMMA GEMM: one wave per 16x16 tile of C = A[MxK] * B
// A tile staged row-major 16x32; B tile staged COLUMN-major (BsT[n][k]) so
// every lane's B fragment is one contiguous 32B LDS run (-> ds_load_b128).
// TRANSB: B(k,n) = Bm[n*ldb + k]   (so staging is coalesced over k)
// EPI==1: C[idx] = Xf[idx] * (Acc[idx] + tile)   (final fused LRP output)
// ---------------------------------------------------------------------------
template <bool TRANSB, int EPI>
__global__ void gemm_wmma(const float* __restrict__ A, int lda,
                          const float* __restrict__ Bm, int ldb,
                          float* __restrict__ C, int ldc,
                          int M, int Ncols, int K,
                          const float* __restrict__ Xf, const float* __restrict__ Acc) {
    __shared__ alignas(32) _Float16 As[16 * 32];
    __shared__ alignas(32) _Float16 BsT[16 * 32];   // [n][k]
    int lane = threadIdx.x;             // blockDim.x == 32
    int m0 = blockIdx.x * 16, n0 = blockIdx.y * 16;
    const v8f vz = {0, 0, 0, 0, 0, 0, 0, 0};
    v8f acc = vz;
    int hiK = (lane >= 16) ? 16 : 0;
    int hiB = (lane >= 16) ? 16 : 0;
    int mrow = lane & 15;
    for (int k0 = 0; k0 < K; k0 += 32) {
        if (k0 + 32 < K)    // prefetch next A panel (global_prefetch_b8)
            __builtin_prefetch(&A[(size_t)(m0 + mrow) * lda + k0 + 32], 0, 3);
        // stage A 16x32 (row-major), f32 -> f16
        for (int i = lane; i < 16 * 32; i += 32) {
            int r = i >> 5, c = i & 31;
            int gk = k0 + c;
            float v = (gk < K) ? A[(size_t)(m0 + r) * lda + gk] : 0.f;
            As[r * 32 + c] = (_Float16)v;
        }
        // stage B 32x16 transposed into BsT[n][k]
        if (TRANSB) {
            for (int i = lane; i < 16 * 32; i += 32) {
                int n = i >> 5, kk = i & 31;        // lanes sweep kk: coalesced
                int gk = k0 + kk;
                float v = (gk < K) ? Bm[(size_t)(n0 + n) * ldb + gk] : 0.f;
                BsT[n * 32 + kk] = (_Float16)v;
            }
        } else {
            for (int i = lane; i < 16 * 32; i += 32) {
                int kk = i >> 4, n = i & 15;        // lanes sweep n: coalesced
                int gk = k0 + kk;
                float v = (gk < K) ? Bm[(size_t)gk * ldb + (n0 + n)] : 0.f;
                BsT[n * 32 + kk] = (_Float16)v;
            }
        }
        __syncthreads();
        const _Float16* ap = &As[mrow * 32 + hiK];
        v16h af = frag16(ap, ap + 8);
        const _Float16* bp = &BsT[mrow * 32 + hiB];
        v16h bf = frag16(bp, bp + 8);
        acc = __builtin_amdgcn_wmma_f32_16x16x32_f16(false, af, false, bf,
                                                     (short)0, acc, false, false);
        __syncthreads();
    }
#pragma unroll
    for (int r = 0; r < 8; ++r) {
        int mm = m0 + r + ((lane >= 16) ? 8 : 0);
        int nn = n0 + mrow;
        if (mm < M && nn < Ncols) {
            size_t idx = (size_t)mm * ldc + nn;
            if (EPI == 0) C[idx] = acc[r];
            else          C[idx] = Xf[idx] * (Acc[idx] + acc[r]);
        }
    }
}

// ---------------------------------------------------------------------------
// Edge predictor forward: one block (128 thr / 4 waves) per edge.
// z = W0 * |h_s (x) h_d|  via WMMA, BN + LeakyReLU, cout, fc1, sigmoid, mean.
// x_ij is staged h-major (xT[h][c]) so B fragments are contiguous LDS runs.
// ---------------------------------------------------------------------------
__global__ void edge_pred_fwd(const float* __restrict__ h,
                              const int* __restrict__ src, const int* __restrict__ dst,
                              const float* __restrict__ w0,
                              const float* __restrict__ bnw, const float* __restrict__ bnb,
                              const float* __restrict__ bnrm, const float* __restrict__ bnrv,
                              const float* __restrict__ coutw, const float* __restrict__ coutb,
                              const float* __restrict__ fc1w, const float* __restrict__ fc1b,
                              float* __restrict__ ep) {
    __shared__ float hs[64], hd[64];
    __shared__ alignas(32) _Float16 w0s[4096];   // [o][c]
    __shared__ alignas(32) _Float16 xT[4096];    // [h][c]
    __shared__ float zb[4096];                   // [o][h]
    __shared__ float sv[64], sig[64];
    int e = blockIdx.x, t = threadIdx.x;
    int s = src[e], d = dst[e];
    if (t < 64) { hs[t] = h[(size_t)s * HID + t]; hd[t] = h[(size_t)d * HID + t]; }
    for (int i = t; i < 4096; i += 128) w0s[i] = (_Float16)w0[i];
    __syncthreads();
    for (int i = t; i < 4096; i += 128) {
        int hc = i >> 6, c = i & 63;
        xT[i] = (_Float16)fabsf(hs[c] - hd[hc]);
    }
    __syncthreads();
    int wave = t >> 5, lane = t & 31;
    const v8f vz = {0, 0, 0, 0, 0, 0, 0, 0};
    v8f acc[4] = {vz, vz, vz, vz};
    int mrow = (wave << 4) + (lane & 15);
    int hiK = (lane >= 16) ? 16 : 0, hiB = (lane >= 16) ? 16 : 0;
    for (int k0 = 0; k0 < 64; k0 += 32) {
        const _Float16* ap = &w0s[mrow * 64 + k0 + hiK];
        v16h af = frag16(ap, ap + 8);
#pragma unroll
        for (int nt = 0; nt < 4; ++nt) {
            const _Float16* bp = &xT[((nt << 4) + (lane & 15)) * 64 + k0 + hiB];
            v16h bf = frag16(bp, bp + 8);
            acc[nt] = __builtin_amdgcn_wmma_f32_16x16x32_f16(false, af, false, bf,
                                                             (short)0, acc[nt], false, false);
        }
    }
#pragma unroll
    for (int nt = 0; nt < 4; ++nt)
#pragma unroll
        for (int r = 0; r < 8; ++r) {
            int o = (wave << 4) + r + ((lane >= 16) ? 8 : 0);
            int hc = (nt << 4) + (lane & 15);
            float scale = bnw[o] * rsqrtf(bnrv[o] + 1e-5f);
            float v = (acc[nt][r] - bnrm[o]) * scale + bnb[o];
            zb[o * 64 + hc] = (v >= 0.f) ? v : 0.01f * v;
        }
    __syncthreads();
    if (t < 64) {
        float a = 0.f;
        for (int o = 0; o < 64; ++o) a += coutw[o] * zb[o * 64 + t];
        sv[t] = a + coutb[0];
    }
    __syncthreads();
    if (t < 64) {
        float a = fc1b[t];
        for (int hc = 0; hc < 64; ++hc) a += sv[hc] * fc1w[t * 64 + hc];
        sig[t] = 1.f / (1.f + __expf(-a));
    }
    __syncthreads();
    if (t == 0) {
        float a = 0.f;
        for (int o = 0; o < 64; ++o) a += sig[o];
        ep[e] = a * (1.f / 64.f);
    }
}

// ---------------------------------------------------------------------------
// Edge predictor backward: recompute forward internals, then two WMMA matmuls
// (z = W0*x_ij and g_xij = W0^T*g_z), scatter |.|-grad into g_h.
// Keeps a transposed W0 copy in LDS so both A fragments are contiguous.
// ---------------------------------------------------------------------------
__global__ void edge_pred_bwd(const float* __restrict__ h,
                              const int* __restrict__ src, const int* __restrict__ dst,
                              const float* __restrict__ w0,
                              const float* __restrict__ bnw, const float* __restrict__ bnb,
                              const float* __restrict__ bnrm, const float* __restrict__ bnrv,
                              const float* __restrict__ coutw, const float* __restrict__ coutb,
                              const float* __restrict__ fc1w, const float* __restrict__ fc1b,
                              const float* __restrict__ g_ep, float* __restrict__ g_h) {
    __shared__ float hs[64], hd[64];
    __shared__ alignas(32) _Float16 w0s[4096];   // [o][c]
    __shared__ alignas(32) _Float16 w0t[4096];   // [c][o]
    __shared__ alignas(32) _Float16 xT[4096];    // [h][c], reused as g_z [h][o]
    __shared__ float zb[4096];                   // z-post [o][h], then g_xij [c][h]
    __shared__ float sv[64], sig[64], gt[64], gsv[64];
    int e = blockIdx.x, t = threadIdx.x;
    int s = src[e], d = dst[e];
    if (t < 64) { hs[t] = h[(size_t)s * HID + t]; hd[t] = h[(size_t)d * HID + t]; }
    for (int i = t; i < 4096; i += 128) {
        int o = i >> 6, c = i & 63;
        _Float16 v = (_Float16)w0[i];
        w0s[i] = v;
        w0t[c * 64 + o] = v;
    }
    __syncthreads();
    for (int i = t; i < 4096; i += 128) {
        int hc = i >> 6, c = i & 63;
        xT[i] = (_Float16)fabsf(hs[c] - hd[hc]);
    }
    __syncthreads();
    int wave = t >> 5, lane = t & 31;
    const v8f vz = {0, 0, 0, 0, 0, 0, 0, 0};
    int mrow = (wave << 4) + (lane & 15);
    int hiK = (lane >= 16) ? 16 : 0, hiB = (lane >= 16) ? 16 : 0;
    {   // forward matmul: z = W0 * x_ij
        v8f acc[4] = {vz, vz, vz, vz};
        for (int k0 = 0; k0 < 64; k0 += 32) {
            const _Float16* ap = &w0s[mrow * 64 + k0 + hiK];
            v16h af = frag16(ap, ap + 8);
#pragma unroll
            for (int nt = 0; nt < 4; ++nt) {
                const _Float16* bp = &xT[((nt << 4) + (lane & 15)) * 64 + k0 + hiB];
                v16h bf = frag16(bp, bp + 8);
                acc[nt] = __builtin_amdgcn_wmma_f32_16x16x32_f16(false, af, false, bf,
                                                                 (short)0, acc[nt], false, false);
            }
        }
#pragma unroll
        for (int nt = 0; nt < 4; ++nt)
#pragma unroll
            for (int r = 0; r < 8; ++r) {
                int o = (wave << 4) + r + ((lane >= 16) ? 8 : 0);
                int hc = (nt << 4) + (lane & 15);
                float scale = bnw[o] * rsqrtf(bnrv[o] + 1e-5f);
                float v = (acc[nt][r] - bnrm[o]) * scale + bnb[o];
                zb[o * 64 + hc] = (v >= 0.f) ? v : 0.01f * v;
            }
    }
    __syncthreads();
    if (t < 64) {
        float a = 0.f;
        for (int o = 0; o < 64; ++o) a += coutw[o] * zb[o * 64 + t];
        sv[t] = a + coutb[0];
    }
    __syncthreads();
    if (t < 64) {
        float a = fc1b[t];
        for (int hc = 0; hc < 64; ++hc) a += sv[hc] * fc1w[t * 64 + hc];
        sig[t] = 1.f / (1.f + __expf(-a));
    }
    __syncthreads();
    float gep = g_ep[e];
    if (t < 64) {
        float sg = sig[t];
        gt[t] = gep * (1.f / 64.f) * sg * (1.f - sg);
    }
    __syncthreads();
    if (t < 64) {
        float a = 0.f;
        for (int o = 0; o < 64; ++o) a += gt[o] * fc1w[o * 64 + t];
        gsv[t] = a;
    }
    __syncthreads();
    // g_z (h-major): xT[h][o] = coutw[o] * g_sv[h] * leakmask(o,h) * bnscale[o]
    for (int i = t; i < 4096; i += 128) {
        int hc = i >> 6, o = i & 63;
        float scale = bnw[o] * rsqrtf(bnrv[o] + 1e-5f);
        float lm = (zb[o * 64 + hc] >= 0.f) ? 1.f : 0.01f;
        xT[i] = (_Float16)(coutw[o] * gsv[hc] * lm * scale);
    }
    __syncthreads();
    {   // backward matmul: g_xij = W0^T * g_z
        v8f acc[4] = {vz, vz, vz, vz};
        for (int k0 = 0; k0 < 64; k0 += 32) {
            const _Float16* ap = &w0t[mrow * 64 + k0 + hiK];
            v16h af = frag16(ap, ap + 8);
#pragma unroll
            for (int nt = 0; nt < 4; ++nt) {
                const _Float16* bp = &xT[((nt << 4) + (lane & 15)) * 64 + k0 + hiB];
                v16h bf = frag16(bp, bp + 8);
                acc[nt] = __builtin_amdgcn_wmma_f32_16x16x32_f16(false, af, false, bf,
                                                                 (short)0, acc[nt], false, false);
            }
        }
        __syncthreads();
#pragma unroll
        for (int nt = 0; nt < 4; ++nt)
#pragma unroll
            for (int r = 0; r < 8; ++r) {
                int c = (wave << 4) + r + ((lane >= 16) ? 8 : 0);
                int hc = (nt << 4) + (lane & 15);
                zb[c * 64 + hc] = acc[nt][r];
            }
    }
    __syncthreads();
    if (t < 64) {
        float gs = 0.f;
        for (int hc = 0; hc < 64; ++hc) gs += zb[t * 64 + hc] * sgnf(hs[t] - hd[hc]);
        atomicAdd(&g_h[(size_t)s * HID + t], gs);
        float gd = 0.f;
        for (int c = 0; c < 64; ++c) gd -= zb[c * 64 + t] * sgnf(hs[c] - hd[t]);
        atomicAdd(&g_h[(size_t)d * HID + t], gd);
    }
}

// ---------------------------------------------------------------------------
// Host launch
// ---------------------------------------------------------------------------
extern "C" void kernel_launch(void* const* d_in, const int* in_sizes, int n_in,
                              void* d_out, int out_size, void* d_ws, size_t ws_size,
                              hipStream_t stream) {
    (void)in_sizes; (void)n_in; (void)out_size; (void)ws_size;
    const float* x      = (const float*)d_in[0];
    const int*   ei     = (const int*)d_in[1];
    const int*   src    = ei;
    const int*   dst    = ei + Ee;
    const int*   batch  = (const int*)d_in[2];
    const int*   root   = (const int*)d_in[3];
    const float* R      = (const float*)d_in[4];
    const float* conv1_w = (const float*)d_in[5];
    const float* conv1_b = (const float*)d_in[6];
    const float* conv2_w = (const float*)d_in[7];
    const float* conv2_b = (const float*)d_in[8];
    const float* sim_c0_w = (const float*)d_in[9];
    const float* sim_bn_w = (const float*)d_in[10];
    const float* sim_bn_b = (const float*)d_in[11];
    const float* sim_bn_rm = (const float*)d_in[12];
    const float* sim_bn_rv = (const float*)d_in[13];
    const float* sim_cout_w = (const float*)d_in[14];
    const float* sim_cout_b = (const float*)d_in[15];
    const float* fc1_w = (const float*)d_in[16];
    const float* fc1_b = (const float*)d_in[17];
    const float* bn1_w = (const float*)d_in[18];
    const float* bn1_b = (const float*)d_in[19];
    const float* bn1_rm = (const float*)d_in[20];
    const float* bn1_rv = (const float*)d_in[21];

    float* ws = (float*)d_ws;
    size_t off = 0;
    auto alloc = [&](size_t n) { float* p = ws + off; off += n; return p; };
    float* dinv1   = alloc(Nn);
    float* h_lin   = alloc((size_t)Nn * HID);      // reused later as gA
    float* hbuf    = alloc((size_t)Nn * HID);
    float* ep      = alloc(Ee);
    float* dinv2   = alloc(Nn);
    float* xc_act  = alloc((size_t)Nn * XC);
    float* hh      = alloc((size_t)Nn * HID);
    float* out_pre = alloc((size_t)Nn * HID);
    float* Zsum    = alloc((size_t)Bb * OC);
    float* cntf    = alloc(Bb);
    float* S       = alloc((size_t)Bb * OC);
    float* g_outpre= alloc((size_t)Nn * HID);
    float* g_hh    = alloc((size_t)Nn * HID);
    float* g_dinv2 = alloc(Nn);
    float* g_ep    = alloc(Ee);
    float* g_h     = alloc((size_t)Nn * HID);
    float* gxc     = alloc((size_t)Nn * XC);
    float* g_x     = alloc((size_t)Nn * FIN);
    float* gA      = h_lin;

    const int TB = 256;
    auto gsz = [](long long n, int tb) { return (unsigned)((n + tb - 1) / tb); };

    // ---- forward: conv1 ----
    fill_k<<<gsz(Nn, TB), TB, 0, stream>>>(dinv1, 1.f, Nn);
    deg_count1<<<gsz(Ee, TB), TB, 0, stream>>>(dst, dinv1, Ee);
    rsqrt_ip<<<gsz(Nn, TB), TB, 0, stream>>>(dinv1, Nn);
    gemm_wmma<true, 0><<<dim3(Nn / 16, HID / 16), 32, 0, stream>>>(
        x, FIN, conv1_w, FIN, h_lin, HID, Nn, HID, FIN, nullptr, nullptr);
    node_init64<<<gsz((long long)Nn * HID, TB), TB, 0, stream>>>(hbuf, h_lin, dinv1, conv1_b, Nn);
    edge_agg<<<gsz((long long)Ee * HID, TB), TB, 0, stream>>>(src, dst, dinv1, nullptr, h_lin, hbuf, Ee);

    // ---- edge predictor forward ----
    edge_pred_fwd<<<Ee, 128, 0, stream>>>(hbuf, src, dst, sim_c0_w, sim_bn_w, sim_bn_b,
                                          sim_bn_rm, sim_bn_rv, sim_cout_w, sim_cout_b,
                                          fc1_w, fc1_b, ep);

    // ---- conv2 forward ----
    fill_k<<<gsz(Nn, TB), TB, 0, stream>>>(dinv2, 1.f, Nn);
    deg_count2<<<gsz(Ee, TB), TB, 0, stream>>>(dst, ep, dinv2, Ee);
    rsqrt_ip<<<gsz(Nn, TB), TB, 0, stream>>>(dinv2, Nn);
    xc_build<<<gsz((long long)Nn * XC, TB), TB, 0, stream>>>(hbuf, x, batch, root,
                                                             bn1_w, bn1_b, bn1_rm, bn1_rv,
                                                             xc_act, Nn);
    gemm_wmma<true, 0><<<dim3(Nn / 16, HID / 16), 32, 0, stream>>>(
        xc_act, XC, conv2_w, XC, hh, HID, Nn, HID, XC, nullptr, nullptr);
    node_init64<<<gsz((long long)Nn * HID, TB), TB, 0, stream>>>(out_pre, hh, dinv2, conv2_b, Nn);
    edge_agg<<<gsz((long long)Ee * HID, TB), TB, 0, stream>>>(src, dst, dinv2, ep, hh, out_pre, Ee);

    // ---- mean pool + S = safe_divide(R, Z) ----
    fill_k<<<gsz((long long)Bb * OC, TB), TB, 0, stream>>>(Zsum, 0.f, (long long)Bb * OC);
    fill_k<<<gsz(Bb, TB), TB, 0, stream>>>(cntf, 0.f, Bb);
    cnt_count<<<gsz(Nn, TB), TB, 0, stream>>>(batch, cntf, Nn);
    z_accum<<<gsz((long long)Nn * OC, TB), TB, 0, stream>>>(out_pre, hbuf, batch, root, Zsum, Nn);
    s_build<<<gsz((long long)Bb * OC, TB), TB, 0, stream>>>(Zsum, cntf, R, S, Bb * OC);

    // ---- backward ----
    fill_k<<<gsz((long long)Nn * HID, TB), TB, 0, stream>>>(g_h, 0.f, (long long)Nn * HID);
    fill_k<<<gsz((long long)Nn * FIN, TB), TB, 0, stream>>>(g_x, 0.f, (long long)Nn * FIN);
    fill_k<<<gsz(Ee, TB), TB, 0, stream>>>(g_ep, 0.f, Ee);

    bwd_mean_split<<<gsz((long long)Nn * OC, TB), TB, 0, stream>>>(out_pre, S, batch, root,
                                                                   cntf, g_outpre, g_h, Nn);
    conv2_bwd_self<<<gsz(Nn, TB), TB, 0, stream>>>(g_outpre, hh, dinv2, g_hh, g_dinv2, Nn);
    conv2_bwd_edge<<<Ee, 64, 0, stream>>>(src, dst, g_outpre, hh, dinv2, ep,
                                          g_ep, g_dinv2, g_hh);
    dinv2deg<<<gsz(Nn, TB), TB, 0, stream>>>(dinv2, g_dinv2, Nn);
    deg2ep<<<gsz(Ee, TB), TB, 0, stream>>>(dst, g_dinv2, g_ep, Ee);
    edge_pred_bwd<<<Ee, 128, 0, stream>>>(hbuf, src, dst, sim_c0_w, sim_bn_w, sim_bn_b,
                                          sim_bn_rm, sim_bn_rv, sim_cout_w, sim_cout_b,
                                          fc1_w, fc1_b, g_ep, g_h);
    gemm_wmma<false, 0><<<dim3(Nn / 16, XC / 16), 32, 0, stream>>>(
        g_hh, HID, conv2_w, XC, gxc, XC, Nn, XC, HID, nullptr, nullptr);
    xc_bwd_split<<<gsz((long long)Nn * XC, TB), TB, 0, stream>>>(gxc, xc_act, bn1_w, bn1_rv,
                                                                 batch, root, g_h, g_x, Nn);
    node_scale64<<<gsz((long long)Nn * HID, TB), TB, 0, stream>>>(g_h, dinv1, gA, Nn);
    edge_agg_rev<<<gsz((long long)Ee * HID, TB), TB, 0, stream>>>(src, dst, dinv1, g_h, gA, Ee);
    // final fused: d_out = x * (g_x + gA @ W1)
    gemm_wmma<false, 1><<<dim3(Nn / 16, FIN / 16), 32, 0, stream>>>(
        gA, HID, conv1_w, FIN, (float*)d_out, FIN, Nn, FIN, HID, x, g_x);
}
